// LSTM_20727512171148
// MI455X (gfx1250) — matrix operand order repaired and checked by reading, not compiled, not run on hardware
//
#include <hip/hip_runtime.h>
#include <hip/hip_bf16.h>

// ---------------------------------------------------------------------------
// Problem constants (from the reference)
// ---------------------------------------------------------------------------
#define BB 64          // batch
#define TT 256         // sequence length
#define HH 1024        // hidden
#define LL 3           // layers
#define ASZ 128        // alphabet
#define GG (4*HH)      // gates width = 4096
#define MM (BB*TT)     // 16384 rows

#define LSTM_USE_TDM 1 // stage W_hh slice into LDS with the Tensor Data Mover

typedef __bf16 bf16;
typedef __attribute__((ext_vector_type(16))) __bf16 v16bf;
typedef __attribute__((ext_vector_type(8)))  __bf16 bf16x8;
typedef __attribute__((ext_vector_type(8)))  float  v8f;
typedef unsigned int u32x4 __attribute__((ext_vector_type(4)));
typedef unsigned int u32x8 __attribute__((ext_vector_type(8)));

// ---------------------------------------------------------------------------
// WMMA helpers (CDNA5 wave32, V_WMMA_F32_16X16X32_BF16)
// ---------------------------------------------------------------------------
__device__ __forceinline__ v8f wmma_bf16(v16bf a, v16bf b, v8f c) {
  return __builtin_amdgcn_wmma_f32_16x16x32_bf16(false, a, false, b,
                                                 (short)0, c, false, false);
}

// A-matrix fragment (16x32 bf16, row-major source with leading dim `ld`).
// ISA layout: lanes 0-15 hold M=lane, K = {0..7, 16..23}; lanes 16-31 hold
// M=lane-16, K = {8..15, 24..31}. Each half is 8 contiguous bf16 = one b128.
__device__ __forceinline__ v16bf load_afrag(const bf16* base, size_t ld) {
  int lane = threadIdx.x & 31;
  int m  = lane & 15;
  int kb = (lane >> 4) * 8;
  union { v16bf v; bf16x8 h[2]; } u;
  const bf16* p = base + (size_t)m * ld + kb;
  u.h[0] = *(const bf16x8*)(p);
  u.h[1] = *(const bf16x8*)(p + 16);
  return u.v;
}

// B-matrix fragment (32x16 bf16) from a [N][K] (transposed) tile with leading
// dim `ld`: lane holds column N=lane&15, K = {0..7,16..23} (lanes 0-15) or
// {8..15,24..31} (lanes 16-31) = two b128 loads.
__device__ __forceinline__ v16bf load_bfrag(const bf16* base, size_t ld) {
  int lane = threadIdx.x & 31;
  int n  = lane & 15;
  int kb = (lane >> 4) * 8;
  union { v16bf v; bf16x8 h[2]; } u;
  const bf16* p = base + (size_t)n * ld + kb;
  u.h[0] = *(const bf16x8*)(p);
  u.h[1] = *(const bf16x8*)(p + 16);
  return u.v;
}

__device__ __forceinline__ float fsigmoid(float x) {
  return 1.0f / (1.0f + __expf(-x));
}

// ---------------------------------------------------------------------------
// Elementwise prep kernels
// ---------------------------------------------------------------------------
__global__ __launch_bounds__(256) void k_cvt_bf16(const float* __restrict__ s,
                                                  bf16* __restrict__ d, int n) {
  int i = blockIdx.x * 256 + threadIdx.x;
  if (i < n) d[i] = (bf16)s[i];
}

__global__ __launch_bounds__(256) void k_bias_sum(const float* __restrict__ a,
                                                  const float* __restrict__ b,
                                                  float* __restrict__ d, int n) {
  int i = blockIdx.x * 256 + threadIdx.x;
  if (i < n) d[i] = a[i] + b[i];
}

// i2h: sequence is exact one-hot -> argmax + row gather. Pure bandwidth op.
__global__ __launch_bounds__(128) void k_i2h(const float* __restrict__ seq,
                                             const float* __restrict__ Wi2h,
                                             const float* __restrict__ bi2h,
                                             bf16* __restrict__ xbf) {
  __shared__ int sidx;
  int bt = blockIdx.x;
  if (threadIdx.x == 0) sidx = 0;
  __syncthreads();
  float s = seq[(size_t)bt * ASZ + threadIdx.x];
  if (s > 0.5f) sidx = threadIdx.x;
  __syncthreads();
  int idx = sidx;
  for (int h = threadIdx.x; h < HH; h += 128)
    xbf[(size_t)bt * HH + h] = (bf16)(Wi2h[(size_t)h * ASZ + idx] + bi2h[h]);
}

// ---------------------------------------------------------------------------
// Generic bf16 GEMM:  C[M,N] = A[M,K] * W[N,K]^T + bias[N]   (fp32 accum/out)
// Block tile 128x128, K-step 32, double-buffered LDS (one barrier per step).
// 8 waves: wave = 32(M) x 64(N) = 2x4 WMMA tiles -> 8 WMMA per stage.
// ---------------------------------------------------------------------------
#define GBM 128
#define GBN 128
#define GBK 32

__global__ __launch_bounds__(256) void k_gemm_bf16(
    const bf16* __restrict__ A, const bf16* __restrict__ W,
    const float* __restrict__ bias, float* __restrict__ C,
    int Mn, int Nn, int Kn) {
  __shared__ bf16 sA[2][GBM * GBK];   // 2 x 8 KB
  __shared__ bf16 sB[2][GBN * GBK];   // 2 x 8 KB, stored [n][k]
  int m0 = blockIdx.y * GBM;
  int n0 = blockIdx.x * GBN;
  int tid  = threadIdx.x;
  int wave = tid >> 5;
  int wm = (wave & 3) * 32;        // wave M offset inside block tile
  int wn = (wave >> 2) * 64;       // wave N offset inside block tile

  // staging coords: each thread moves 32 B of A and 32 B of W per stage
  int sr = tid >> 1, sc = (tid & 1) * 16;

  v8f acc[2][4];
  for (int i = 0; i < 2; i++)
    for (int j = 0; j < 4; j++)
      acc[i][j] = (v8f){0.f,0.f,0.f,0.f,0.f,0.f,0.f,0.f};

  auto stage = [&](int k0, int buf) {
    const bf16* ga = A + (size_t)(m0 + sr) * Kn + k0 + sc;
    *(bf16x8*)(&sA[buf][sr * GBK + sc])     = *(const bf16x8*)(ga);
    *(bf16x8*)(&sA[buf][sr * GBK + sc + 8]) = *(const bf16x8*)(ga + 8);
    const bf16* gb = W + (size_t)(n0 + sr) * Kn + k0 + sc;
    *(bf16x8*)(&sB[buf][sr * GBK + sc])     = *(const bf16x8*)(gb);
    *(bf16x8*)(&sB[buf][sr * GBK + sc + 8]) = *(const bf16x8*)(gb + 8);
  };

  stage(0, 0);
  __syncthreads();
  int nk = Kn / GBK;
  for (int kt = 0; kt < nk; kt++) {
    int cur = kt & 1;
    if (kt + 1 < nk) stage((kt + 1) * GBK, cur ^ 1);  // overlaps WMMA below
    v16bf a0 = load_afrag(&sA[cur][(size_t)(wm +  0) * GBK], GBK);
    v16bf a1 = load_afrag(&sA[cur][(size_t)(wm + 16) * GBK], GBK);
    for (int j = 0; j < 4; j++) {
      v16bf b = load_bfrag(&sB[cur][(size_t)(wn + j * 16) * GBK], GBK);
      acc[0][j] = wmma_bf16(a0, b, acc[0][j]);
      acc[1][j] = wmma_bf16(a1, b, acc[1][j]);
    }
    __syncthreads();
  }

  // Epilogue: C/D layout — VGPR r holds M=r (lanes 0-15) / M=8+r (lanes 16-31)
  int lane = tid & 31;
  int cn = lane & 15;
  int mb = (lane >> 4) * 8;
  for (int i = 0; i < 2; i++)
    for (int j = 0; j < 4; j++) {
      int col = n0 + wn + j * 16 + cn;
      float bv = bias[col];
      for (int r = 0; r < 8; r++) {
        int row = m0 + wm + i * 16 + mb + r;
        C[(size_t)row * Nn + col] = acc[i][j][r] + bv;
      }
    }
}

// ---------------------------------------------------------------------------
// Persistent cooperative LSTM recurrence (one launch per layer).
// 64 blocks; block b owns hidden cols [16b,16b+16) -> 64 gate columns
// {q*H + hc0 + j}. Its W_hh slice (64x1024 bf16 = 128 KB) lives in LDS for
// the whole layer (staged by the Tensor Data Mover). c-state lives in
// registers. One grid barrier per step.
// ---------------------------------------------------------------------------
#define RNB 64
#define RSMEM (64*1024*2 + 64*64*4)   // 128 KB weights + 16 KB gate exchange

__device__ __forceinline__ void grid_barrier(unsigned* bar, unsigned target) {
  __syncthreads();
  if (threadIdx.x == 0) {
    __threadfence();
    atomicAdd(bar, 1u);
    while (__hip_atomic_load(bar, __ATOMIC_ACQUIRE, __HIP_MEMORY_SCOPE_AGENT) <
           target)
      __builtin_amdgcn_s_sleep(1);
  }
  __syncthreads();
}

__global__ __launch_bounds__(256) void k_lstm_recur(
    const float* __restrict__ xg,     // (B*T, G) fp32, biases folded in
    const bf16*  __restrict__ Whh,    // (G, H) bf16, this layer
    const float* __restrict__ h0,     // (B, H) this layer
    const float* __restrict__ c0,     // (B, H)
    bf16* __restrict__ hinit,         // (B, H) bf16 scratch (t=0 input)
    bf16* __restrict__ hs,            // (B, T, H) bf16 layer output sequence
    float* __restrict__ hT,           // (B, H) slice of d_out h_fin
    float* __restrict__ cT,           // (B, H) slice of d_out c_fin
    unsigned* __restrict__ bar) {
  extern __shared__ char smem[];
  bf16*  sW = (bf16*)smem;                       // [64][1024]
  float* sG = (float*)(smem + 64 * 1024 * 2);    // [64][64]

  int hc0 = blockIdx.x * 16;
  int tid = threadIdx.x;
  int wave = tid >> 5, lane = tid & 31;

#if LSTM_USE_TDM
  // --- Tensor Data Mover: view W_hh (4096x1024 bf16) as a 3-D tensor
  // [quadrant=4][row=1024][col=1024] and pull tile [4][16 rows @ hc0][1024]
  // straight into LDS (contiguous in tile-iteration order == sW[q*16+j][k]).
  if (wave == 0) {
    // LDS aperture keeps the byte offset in addr[31:0] (ISA 10.2).
    unsigned lds_addr = (unsigned)(size_t)sW;
    unsigned long long ga =
        (unsigned long long)(size_t)Whh + (size_t)hc0 * HH * 2;
    u32x4 g0 = { 0x1u,                       // count=1, user descriptor
                 lds_addr,                   // D#.lds_addr
                 (unsigned)ga,               // global_addr[31:0]
                 (unsigned)((ga >> 32) & 0x01FFFFFFu) | 0x80000000u }; // +type=2
    u32x8 g1 = { 0x00010000u,   // workgroup_mask=0, data_size=1 (2 bytes)
                 0x04000000u,   // tensor_dim0 = 1024 (bits 79:48, low half)
                 0x04000000u,   // tensor_dim0 hi=0 | tensor_dim1=1024 lo
                 0x04000000u,   // tensor_dim1 hi=0 | tile_dim0=1024
                 0x00040010u,   // tile_dim1=16 | tile_dim2=4
                 0x00000400u,   // tensor_dim0_stride = 1024
                 0x00000000u,   // stride0 hi | tensor_dim1_stride lo16 (2^20)
                 0x00000010u }; // tensor_dim1_stride[47:16] = 0x10
    u32x4 g2 = { 4u, 1u, 0u, 0u };  // tensor_dim2=4, tensor_dim3=1, tile_dim3=0
    u32x4 g3 = { 0u, 0u, 0u, 0u };
    asm volatile("tensor_load_to_lds %0, %1, %2, %3"
                 :: "s"(g0), "s"(g1), "s"(g2), "s"(g3)
                 : "memory");
    __builtin_amdgcn_s_wait_tensorcnt(0);
  }
#else
  for (int i = tid; i < 64 * (HH / 8); i += 256) {
    int nl = i >> 7;                // local gate row 0..63
    int kc = (i & 127) * 8;         // k offset
    int g  = (nl >> 4) * HH + hc0 + (nl & 15);
    *(bf16x8*)(sW + (size_t)nl * HH + kc) =
        *(const bf16x8*)(Whh + (size_t)g * HH + kc);
  }
#endif

  // Register-resident c state; publish bf16 h0 for step 0.
  float creg[4];
  for (int e = 0; e < 4; e++) {
    int lin = tid + 256 * e;        // 1024 = 64 m x 16 j
    int m = lin >> 4, j = lin & 15;
    creg[e] = c0[(size_t)m * HH + hc0 + j];
    hinit[(size_t)m * HH + hc0 + j] = (bf16)h0[(size_t)m * HH + hc0 + j];
  }
  grid_barrier(bar, RNB * 1u);      // hinit visible everywhere; sW loaded

  int wn  = (wave & 3) * 16;        // local gate-col tile (quadrant = wn>>4)
  int wmb = (wave >> 2) * 32;       // batch-row offset (0 or 32)
  int cn  = lane & 15;
  int mb  = (lane >> 4) * 8;
  int gcol = (wn >> 4) * HH + hc0 + cn;

  for (int t = 0; t < TT; t++) {
    const bf16* hbase;  size_t ldA;
    if (t == 0) { hbase = hinit;                    ldA = HH; }
    else        { hbase = hs + (size_t)(t - 1) * HH; ldA = (size_t)TT * HH; }

    // acc preloaded with xg_t (biases already folded in)
    v8f acc[2];
    for (int i = 0; i < 2; i++)
      for (int r = 0; r < 8; r++) {
        int m = wmb + i * 16 + mb + r;
        acc[i][r] = xg[((size_t)m * TT + t) * GG + gcol];
      }

    for (int k0 = 0; k0 < HH; k0 += 32) {
      v16bf bfr = load_bfrag(sW + (size_t)wn * HH + k0, HH);
      v16bf a0  = load_afrag(hbase + (size_t)(wmb +  0) * ldA + k0, ldA);
      v16bf a1  = load_afrag(hbase + (size_t)(wmb + 16) * ldA + k0, ldA);
      acc[0] = wmma_bf16(a0, bfr, acc[0]);
      acc[1] = wmma_bf16(a1, bfr, acc[1]);
    }

    // Exchange gates through LDS so one thread owns all 4 gates of (m,j).
    for (int i = 0; i < 2; i++)
      for (int r = 0; r < 8; r++)
        sG[(size_t)(wmb + i * 16 + mb + r) * 64 + wn + cn] = acc[i][r];
    __syncthreads();

    for (int e = 0; e < 4; e++) {
      int lin = tid + 256 * e;
      int m = lin >> 4, j = lin & 15;
      float gi = sG[(size_t)m * 64 +      j];
      float gf = sG[(size_t)m * 64 + 16 + j];
      float gg = sG[(size_t)m * 64 + 32 + j];
      float go = sG[(size_t)m * 64 + 48 + j];
      creg[e] = fsigmoid(gf) * creg[e] + fsigmoid(gi) * tanhf(gg);
      float h = fsigmoid(go) * tanhf(creg[e]);
      hs[((size_t)m * TT + t) * HH + hc0 + j] = (bf16)h;
      if (t + 1 < TT) {   // prefetch next step's gate inputs (global_prefetch_b8)
        __builtin_prefetch(&xg[((size_t)m * TT + t + 1) * GG + hc0 + j], 0, 1);
      } else {
        hT[(size_t)m * HH + hc0 + j] = h;
        cT[(size_t)m * HH + hc0 + j] = creg[e];
      }
    }
    if (t < TT - 1) grid_barrier(bar, RNB * (unsigned)(t + 2));
  }
}

// ---------------------------------------------------------------------------
// Host-side orchestration
// ---------------------------------------------------------------------------
extern "C" void kernel_launch(void* const* d_in, const int* in_sizes, int n_in,
                              void* d_out, int out_size, void* d_ws,
                              size_t ws_size, hipStream_t stream) {
  (void)in_sizes; (void)n_in; (void)out_size; (void)ws_size;
  const float* seq  = (const float*)d_in[0];
  const float* Wi2h = (const float*)d_in[1];
  const float* bi2h = (const float*)d_in[2];
  const float* Wih  = (const float*)d_in[3];
  const float* Whh  = (const float*)d_in[4];
  const float* bih  = (const float*)d_in[5];
  const float* bhh  = (const float*)d_in[6];
  const float* Wh2o = (const float*)d_in[7];
  const float* bh2o = (const float*)d_in[8];
  const float* h0   = (const float*)d_in[9];
  const float* c0   = (const float*)d_in[10];
  float* out = (float*)d_out;

  // Workspace carve-up (~420 MB)
  char* p = (char*)d_ws;
  auto take = [&](size_t bytes) -> void* {
    void* r = (void*)p;
    p += (bytes + 255) & ~(size_t)255;
    return r;
  };
  bf16*  x_bf   = (bf16*) take((size_t)MM * HH * 2);
  bf16*  hsA    = (bf16*) take((size_t)MM * HH * 2);
  bf16*  hsB    = (bf16*) take((size_t)MM * HH * 2);
  float* xg     = (float*)take((size_t)MM * GG * 4);
  bf16*  wih_bf = (bf16*) take((size_t)LL * GG * HH * 2);
  bf16*  whh_bf = (bf16*) take((size_t)LL * GG * HH * 2);
  bf16*  wh2o_bf= (bf16*) take((size_t)ASZ * HH * 2);
  float* bsum   = (float*)take((size_t)LL * GG * 4);
  bf16*  hinit  = (bf16*) take((size_t)BB * HH * 2);
  unsigned* bar = (unsigned*)take(256);

  // --- prep: bf16 weight copies + folded biases ---
  {
    int n = LL * GG * HH;
    k_cvt_bf16<<<(n + 255) / 256, 256, 0, stream>>>(Wih, wih_bf, n);
    k_cvt_bf16<<<(n + 255) / 256, 256, 0, stream>>>(Whh, whh_bf, n);
    int n2 = ASZ * HH;
    k_cvt_bf16<<<(n2 + 255) / 256, 256, 0, stream>>>(Wh2o, wh2o_bf, n2);
    int n3 = LL * GG;
    k_bias_sum<<<(n3 + 255) / 256, 256, 0, stream>>>(bih, bhh, bsum, n3);
  }

  // --- i2h gather (one-hot) ---
  k_i2h<<<MM, 128, 0, stream>>>(seq, Wi2h, bi2h, x_bf);

  // --- h2O applied to x (faithful to the reference's bug) ---
  k_gemm_bf16<<<dim3(ASZ / GBN, MM / GBM), 256, 0, stream>>>(
      x_bf, wh2o_bf, bh2o, out, MM, ASZ, HH);

  // --- 3-layer LSTM ---
  float* hfin = out + (size_t)MM * ASZ;
  float* cfin = hfin + (size_t)LL * BB * HH;
  const bf16* layer_in = x_bf;
  bf16* outA = hsA;
  bf16* outB = hsB;
  for (int l = 0; l < LL; l++) {
    // xg = layer_in @ W_ih[l]^T + (b_ih[l] + b_hh[l])
    k_gemm_bf16<<<dim3(GG / GBN, MM / GBM), 256, 0, stream>>>(
        layer_in, wih_bf + (size_t)l * GG * HH, bsum + (size_t)l * GG, xg,
        MM, GG, HH);
    hipMemsetAsync(bar, 0, 256, stream);
    k_lstm_recur<<<RNB, 256, RSMEM, stream>>>(
        xg, whh_bf + (size_t)l * GG * HH,
        h0 + (size_t)l * BB * HH, c0 + (size_t)l * BB * HH,
        hinit, outA,
        hfin + (size_t)l * BB * HH, cfin + (size_t)l * BB * HH, bar);
    layer_in = outA;
    bf16* tmp = outA; outA = outB; outB = tmp;
  }
}
